// RiceAttention_21912923144247
// MI455X (gfx1250) — compile-verified, matrix operand order checked by reading, not run on hardware
//
#include <hip/hip_runtime.h>
#include <hip/hip_bf16.h>

typedef __attribute__((ext_vector_type(16))) _Float16 v16h;
typedef __attribute__((ext_vector_type(8)))  _Float16 v8h;
typedef __attribute__((ext_vector_type(8)))  float    v8f;

#define S_LEN 4096
#define DIM   1280
#define NH    16
#define HD    80
#define HDP   96   // padded head dim for QK^T (3 x 32 WMMA K-steps)

static __device__ __forceinline__ v8f wmma_f16(v16h a, v16h b, v8f c) {
    // D = A(16x32 f16) x B(32x16 f16) + C(16x16 f32)
    return __builtin_amdgcn_wmma_f32_16x16x32_f16(
        /*neg_a=*/false, a, /*neg_b=*/false, b,
        /*c_mod=*/(short)0, c, /*reuse_a=*/false, /*reuse_b=*/false);
}

// ---------------------------------------------------------------------------
// Tiled GEMM with bias: out[M,N] = A[M,K] * W[N,K]^T + bias[N]
// Block tile 128x64, 256 threads = 8 waves (4x2), each wave a 32x32 tile:
// 4 x v_wmma_f32_16x16x32_f16 per 32-deep K step.
// A staged to LDS as f16 (gfx1250 async b128 copy when already f16).
// ---------------------------------------------------------------------------
template<bool A_HALF, bool OUT_HALF>
__global__ __launch_bounds__(256) void gemm_bias_kernel(
    const void* __restrict__ Ap, const float* __restrict__ W,
    const float* __restrict__ bias, void* __restrict__ outp,
    int M, int N, int K)
{
    __shared__ alignas(32) _Float16 As[128][40];
    __shared__ alignas(32) _Float16 Bs[64][40];

    const int t    = threadIdx.x;
    const int m0   = blockIdx.y * 128;
    const int n0   = blockIdx.x * 64;
    const int lane = t & 31;
    const int wave = t >> 5;
    const int wr   = wave & 3;     // 4 row waves (32 rows each)
    const int wc   = wave >> 2;    // 2 col waves (32 cols each)
    const int l16  = lane & 15;
    const int hlf  = lane >> 4;
    const int ar   = t >> 1;          // A staging row 0..127
    const int ac   = (t & 1) * 16;    // A staging col 0/16
    const int br   = t >> 2;          // B staging row 0..63
    const int bc   = (t & 3) * 8;     // B staging col 0,8,16,24

    v8f acc[2][2];
#pragma unroll
    for (int i = 0; i < 2; ++i)
#pragma unroll
        for (int j = 0; j < 2; ++j) acc[i][j] = (v8f)0.0f;

    for (int k0 = 0; k0 < K; k0 += 32) {
        // ---- stage A tile ----
        if constexpr (A_HALF) {
            const _Float16* A = (const _Float16*)Ap;
            const _Float16* gp = A + (size_t)(m0 + ar) * K + k0 + ac;
            unsigned lds0 = (unsigned)(size_t)&As[ar][ac];
            unsigned lds1 = (unsigned)(size_t)&As[ar][ac + 8];
            unsigned long long g0 = (unsigned long long)(size_t)gp;
            unsigned long long g1 = g0 + 16ull;
            // gfx1250 async copy global -> LDS (ASYNCcnt-tracked, no VGPRs)
            asm volatile("global_load_async_to_lds_b128 %0, %1, off"
                         :: "v"(lds0), "v"(g0) : "memory");
            asm volatile("global_load_async_to_lds_b128 %0, %1, off"
                         :: "v"(lds1), "v"(g1) : "memory");
        } else {
            const float* A = (const float*)Ap;
            const float* p = A + (size_t)(m0 + ar) * K + k0 + ac;
            if (k0 + 32 < K) __builtin_prefetch(p + 32, 0, 1);
#pragma unroll
            for (int j = 0; j < 16; ++j) As[ar][ac + j] = (_Float16)p[j];
        }
        // ---- stage W tile (fp32 -> f16) ----
        {
            const float* p = W + (size_t)(n0 + br) * K + k0 + bc;
            if (k0 + 32 < K) __builtin_prefetch(p + 32, 0, 1);
#pragma unroll
            for (int j = 0; j < 8; ++j) Bs[br][bc + j] = (_Float16)p[j];
        }
        if constexpr (A_HALF) {
            asm volatile("s_wait_asynccnt 0" ::: "memory");
        }
        __syncthreads();

        v16h af[2], bf[2];
#pragma unroll
        for (int i = 0; i < 2; ++i)
            af[i] = *(const v16h*)&As[wr * 32 + i * 16 + l16][hlf * 16];
#pragma unroll
        for (int j = 0; j < 2; ++j)
            bf[j] = *(const v16h*)&Bs[wc * 32 + j * 16 + l16][hlf * 16];
#pragma unroll
        for (int i = 0; i < 2; ++i)
#pragma unroll
            for (int j = 0; j < 2; ++j)
                acc[i][j] = wmma_f16(af[i], bf[j], acc[i][j]);
        __syncthreads();
    }

    // ---- epilogue: C layout lane L -> N = L%16, rows = i + 8*(L/16) ----
#pragma unroll
    for (int am = 0; am < 2; ++am) {
#pragma unroll
        for (int bn = 0; bn < 2; ++bn) {
            const int n = n0 + wc * 32 + bn * 16 + l16;
            const float b = bias[n];
#pragma unroll
            for (int i = 0; i < 8; ++i) {
                const int m = m0 + wr * 32 + am * 16 + i + 8 * hlf;
                const float v = acc[am][bn][i] + b;
                if constexpr (OUT_HALF)
                    ((_Float16*)outp)[(size_t)m * N + n] = (_Float16)v;
                else
                    ((float*)outp)[(size_t)m * N + n] = v;
            }
        }
    }
}

// ---------------------------------------------------------------------------
// Segment-id table: segTab[s] = searchsorted(cu[1:], s, right)
// ---------------------------------------------------------------------------
__global__ __launch_bounds__(256) void seg_kernel(const int* __restrict__ cu,
                                                  int* __restrict__ segTab)
{
    const int s = blockIdx.x * 256 + threadIdx.x;
    if (s >= S_LEN) return;
    int g = 0;
#pragma unroll
    for (int j = 1; j <= 8; ++j) g += (s >= cu[j]) ? 1 : 0;
    segTab[s] = g;
}

// ---------------------------------------------------------------------------
// RoPE + head-major scatter.  qkv16: [S][3*DIM] f16 (q|k|v concatenated).
// Q,K -> [NH][S][HDP] f16 (dims 80..95 zeroed; Q pre-scaled by 1/sqrt(80)).
// V   -> transposed Vt[NH][HD][S] f16 (so PV B-fragments load contiguously).
// ---------------------------------------------------------------------------
__global__ __launch_bounds__(256) void rope_kernel(
    const _Float16* __restrict__ qkv, const float* __restrict__ cosb,
    const float* __restrict__ sinb, _Float16* __restrict__ Q,
    _Float16* __restrict__ Kh, _Float16* __restrict__ Vt)
{
    const int idx = blockIdx.x * blockDim.x + threadIdx.x;
    if (idx >= S_LEN * NH * HDP) return;
    const int d = idx % HDP;
    const int t = idx / HDP;
    const int h = t % NH;
    const int s = t / NH;
    const size_t qko = (size_t)(h * S_LEN + s) * HDP + d;
    if (d >= HD) { Q[qko] = (_Float16)0.f; Kh[qko] = (_Float16)0.f; return; }

    const float c  = cosb[s * HD + d];
    const float sn = sinb[s * HD + d];
    const size_t rb = (size_t)s * (3 * DIM);
    const int col = h * HD + d;
    const int d2  = (d < (HD / 2)) ? d + HD / 2 : d - HD / 2;
    const float sgn = (d < (HD / 2)) ? -1.f : 1.f;

    const float qv = (float)qkv[rb + col];
    const float qp = (float)qkv[rb + h * HD + d2];
    const float kv = (float)qkv[rb + DIM + col];
    const float kp = (float)qkv[rb + DIM + h * HD + d2];
    const float vv = (float)qkv[rb + 2 * DIM + col];

    const float inv_sqrt_hd = 0.11180339887498949f; // 1/sqrt(80)
    Q[qko]  = (_Float16)((qv * c + sgn * qp * sn) * inv_sqrt_hd);
    Kh[qko] = (_Float16)(kv * c + sgn * kp * sn);
    Vt[(size_t)(h * HD + d) * S_LEN + s] = (_Float16)vv;
}

// ---------------------------------------------------------------------------
// Online-softmax update for one 16x32 score tile.  MASKED=false is the fast
// path for key tiles entirely inside the (single) segment of all 16 queries.
// ---------------------------------------------------------------------------
template<bool MASKED>
static __device__ __forceinline__ void softmax_update(
    const v8f& c0, const v8f& c1, const int* qseg, int ks0, int ks1,
    float* mrow, float* lrow, float* fs, _Float16 (*sP)[40],
    int l16, int hlf)
{
#pragma unroll
    for (int i = 0; i < 8; ++i) {
        const int row = i + 8 * hlf;
        bool v0 = true, v1 = true;
        float s0 = c0[i], s1 = c1[i];
        if constexpr (MASKED) {
            const int qs = qseg[row];
            v0 = (qs == ks0);
            v1 = (qs == ks1);
            s0 = v0 ? s0 : -1e30f;
            s1 = v1 ? s1 : -1e30f;
        }
        float tmax = fmaxf(s0, s1);
#pragma unroll
        for (int mdl = 1; mdl < 16; mdl <<= 1)
            tmax = fmaxf(tmax, __shfl_xor(tmax, mdl, 16));
        const float mnew = fmaxf(mrow[i], tmax);
        const float f = __expf(mrow[i] - mnew);
        float p0, p1;
        if constexpr (MASKED) {
            p0 = v0 ? __expf(c0[i] - mnew) : 0.f;
            p1 = v1 ? __expf(c1[i] - mnew) : 0.f;
        } else {
            p0 = __expf(c0[i] - mnew);
            p1 = __expf(c1[i] - mnew);
        }
        float rs = p0 + p1;
#pragma unroll
        for (int mdl = 1; mdl < 16; mdl <<= 1)
            rs += __shfl_xor(rs, mdl, 16);
        lrow[i] = lrow[i] * f + rs;
        mrow[i] = mnew;
        fs[i]   = f;
        sP[row][l16]      = (_Float16)p0;
        sP[row][16 + l16] = (_Float16)p1;
    }
}

// ---------------------------------------------------------------------------
// Flash attention over block-diagonal segments.  One wave per (head, 16-query
// tile).  Keys iterated in tiles of 32 bounded by the tile's segment range;
// cross-segment entries masked by select (EXEC stays all-ones for WMMA).
// ---------------------------------------------------------------------------
__global__ __launch_bounds__(32) void attn_kernel(
    const _Float16* __restrict__ Q, const _Float16* __restrict__ Kh,
    const _Float16* __restrict__ Vt, const int* __restrict__ cu,
    const int* __restrict__ segTab, _Float16* __restrict__ O)
{
    const int qt   = blockIdx.x;
    const int h    = blockIdx.y;
    const int lane = threadIdx.x;
    const int l16  = lane & 15;
    const int hlf  = lane >> 4;
    const int q0   = qt * 16;

    __shared__ int qseg[16];
    __shared__ alignas(32) _Float16 sP[16][40];
    if (lane < 16) qseg[lane] = segTab[q0 + lane];
    __syncthreads();

    // Q fragments (A layout): lane -> row q0 + L%16, halves select K range.
    v16h qa[3];
    {
        const _Float16* qb = Q + ((size_t)(h * S_LEN + q0 + l16)) * HDP + hlf * 16;
#pragma unroll
        for (int kk = 0; kk < 3; ++kk) qa[kk] = *(const v16h*)(qb + kk * 32);
    }

    v8f o[5];
#pragma unroll
    for (int j = 0; j < 5; ++j) o[j] = (v8f)0.0f;
    float mrow[8], lrow[8], fs[8];
#pragma unroll
    for (int i = 0; i < 8; ++i) { mrow[i] = -1e30f; lrow[i] = 0.f; }

    const int sgLo = segTab[q0];
    const int sgHi = segTab[q0 + 15];
    const int klo  = cu[sgLo] & ~31;
    const int khi  = cu[sgHi + 1];
    const bool qUniform = (sgLo == sgHi);

    for (int kb = klo; kb < khi; kb += 32) {
        // ---- S = Q K^T for 32 keys (two 16-key WMMA column tiles) ----
        v8f c0 = (v8f)0.0f, c1 = (v8f)0.0f;
        const _Float16* kb0 = Kh + ((size_t)(h * S_LEN + kb + l16)) * HDP + hlf * 16;
        const _Float16* kb1 = kb0 + (size_t)16 * HDP;
#pragma unroll
        for (int kk = 0; kk < 3; ++kk) {
            c0 = wmma_f16(qa[kk], *(const v16h*)(kb0 + kk * 32), c0);
            c1 = wmma_f16(qa[kk], *(const v16h*)(kb1 + kk * 32), c1);
        }

        // Fast path: tile entirely inside the single segment of all queries.
        // Condition is wave-uniform, so EXEC stays all-ones either way.
        const int ktLo = segTab[kb];
        const int ktHi = segTab[kb + 31];
        if (qUniform && ktLo == ktHi && ktLo == sgLo) {
            softmax_update<false>(c0, c1, qseg, 0, 0, mrow, lrow, fs, sP, l16, hlf);
        } else {
            const int ks0 = segTab[kb + l16];
            const int ks1 = segTab[kb + 16 + l16];
            softmax_update<true>(c0, c1, qseg, ks0, ks1, mrow, lrow, fs, sP, l16, hlf);
        }
        __syncthreads();

        // ---- reload P in A-fragment layout; accumulate O += P * V ----
        v16h pf = *(const v16h*)&sP[l16][hlf * 16];
#pragma unroll
        for (int j = 0; j < 5; ++j) {
            v16h bv = *(const v16h*)(Vt + ((size_t)(h * HD + j * 16 + l16)) * S_LEN
                                        + kb + hlf * 16);
#pragma unroll
            for (int i = 0; i < 8; ++i) o[j][i] *= fs[i];
            o[j] = wmma_f16(pf, bv, o[j]);
        }
        __syncthreads();
    }

    // ---- normalize and store (f16, row-major [S][DIM], head-major cols) ----
    float inv[8];
#pragma unroll
    for (int i = 0; i < 8; ++i) inv[i] = 1.0f / lrow[i];
#pragma unroll
    for (int j = 0; j < 5; ++j) {
#pragma unroll
        for (int i = 0; i < 8; ++i) {
            const int row = i + 8 * hlf;
            const float v = o[j][i] * inv[i];
            O[(size_t)(q0 + row) * DIM + h * HD + j * 16 + l16] = (_Float16)v;
        }
    }
}

// ---------------------------------------------------------------------------
extern "C" void kernel_launch(void* const* d_in, const int* in_sizes, int n_in,
                              void* d_out, int out_size, void* d_ws, size_t ws_size,
                              hipStream_t stream) {
    const float* hidden = (const float*)d_in[0];   // [4096][1280]
    const float* cosb   = (const float*)d_in[1];   // [4096][80]
    const float* sinb   = (const float*)d_in[2];   // [4096][80]
    const float* qkv_w  = (const float*)d_in[3];   // [3840][1280]
    const float* qkv_b  = (const float*)d_in[4];   // [3840]
    const float* proj_w = (const float*)d_in[5];   // [1280][1280]
    const float* proj_b = (const float*)d_in[6];   // [1280]
    const int*   cu     = (const int*)d_in[7];     // [9]
    float* out = (float*)d_out;                    // [4096][1280] fp32

    // workspace carve-up
    char* ws = (char*)d_ws;
    _Float16* qkv16 = (_Float16*)ws;                                   // S*3DIM
    _Float16* Qh    = (_Float16*)(ws + (size_t)S_LEN * 3 * DIM * 2);   // NH*S*HDP
    _Float16* Kh    = Qh + (size_t)NH * S_LEN * HDP;
    _Float16* Vt    = Kh + (size_t)NH * S_LEN * HDP;                   // NH*HD*S
    _Float16* attn  = Vt + (size_t)NH * HD * S_LEN;                    // S*DIM
    int*      segT  = (int*)(attn + (size_t)S_LEN * DIM);              // S

    // 0) segment-id table
    seg_kernel<<<(S_LEN + 255) / 256, 256, 0, stream>>>(cu, segT);

    // 1) QKV projection: [4096,3840] = hidden @ qkv_w^T + qkv_b  (f16 out)
    gemm_bias_kernel<false, true><<<dim3(3 * DIM / 64, S_LEN / 128), 256, 0, stream>>>(
        hidden, qkv_w, qkv_b, qkv16, S_LEN, 3 * DIM, DIM);

    // 2) RoPE + head-major scatter (Q scaled by 1/sqrt(HD), V transposed)
    {
        int total = S_LEN * NH * HDP;
        rope_kernel<<<(total + 255) / 256, 256, 0, stream>>>(
            qkv16, cosb, sinb, Qh, Kh, Vt);
    }

    // 3) Block-diagonal flash attention
    attn_kernel<<<dim3(S_LEN / 16, NH), 32, 0, stream>>>(Qh, Kh, Vt, cu, segT, attn);

    // 4) Output projection: out = attn @ proj_w^T + proj_b  (fp32 out)
    gemm_bias_kernel<true, false><<<dim3(DIM / 64, S_LEN / 128), 256, 0, stream>>>(
        attn, proj_w, proj_b, out, S_LEN, DIM, DIM);
}
